// ClawMatrix_4647154614576
// MI455X (gfx1250) — matrix operand order repaired and checked by reading, not compiled, not run on hardware
//
#include <hip/hip_runtime.h>

// ---------------------------------------------------------------------------
// CDNA5 / gfx1250 fused ClawMatrix pipeline.
// All six GEMMs run on v_wmma_f32_16x16x32_bf16 with LDS-free, stride-1
// operand fragments (A row-major bf16, B stored pre-transposed [N][K]).
// ---------------------------------------------------------------------------

typedef __attribute__((ext_vector_type(16))) __bf16 v16bf;
typedef __attribute__((ext_vector_type(8)))  float  v8f;

struct U32x8 { uint4 lo; uint4 hi; };  // 32 bytes == one v16bf fragment

static __device__ __forceinline__ unsigned short f2bf(float f) {
    unsigned int u = __float_as_uint(f);
    return (unsigned short)((u + 0x7FFFu + ((u >> 16) & 1u)) >> 16);
}

// A-fragment (16x32 bf16): lane=(lr,khalf); elements e -> k = e + 16*khalf.
// Two contiguous 16B loads per lane.
static __device__ __forceinline__ v16bf a_frag(const unsigned short* arow) {
    U32x8 s;
    s.lo = *(const uint4*)arow;          // k = 16*khalf .. +7
    s.hi = *(const uint4*)(arow + 8);    // k = 16*khalf+8 .. +7
    return __builtin_bit_cast(v16bf, s);
}

// B-fragment (32x16 bf16): lane lr = column, elements e -> k = e + 16*khalf.
// One contiguous 32B load per lane (B stored transposed, k contiguous).
static __device__ __forceinline__ v16bf b_frag(const unsigned short* p) {
    const uint4* q = (const uint4*)p;
    U32x8 s; s.lo = q[0]; s.hi = q[1];
    return __builtin_bit_cast(v16bf, s);
}

static __device__ __forceinline__ v8f wmma_bf16(v16bf a, v16bf b, v8f c) {
    return __builtin_amdgcn_wmma_f32_16x16x32_bf16(false, a, false, b, (short)0, c, false, false);
}

// ---------------------------------------------------------------------------
// scale = mean(claw) / TEMPERATURE
// ---------------------------------------------------------------------------
__global__ __launch_bounds__(1024) void claw_scale_kernel(const float* __restrict__ claw,
                                                          float* __restrict__ outScale) {
    __shared__ float red[1024];
    int tid = threadIdx.x;
    float s = 0.f;
    for (int i = tid; i < 768 * 768; i += 1024) s += claw[i];
    red[tid] = s;
    __syncthreads();
    for (int w = 512; w > 0; w >>= 1) {
        if (tid < w) red[tid] += red[tid + w];
        __syncthreads();
    }
    if (tid == 0) outScale[0] = red[0] / (768.f * 768.f) / 0.07f;
}

// ---------------------------------------------------------------------------
// f32 -> bf16, row-major copy
// ---------------------------------------------------------------------------
__global__ __launch_bounds__(256) void cvt_bf16_kernel(const float* __restrict__ src,
                                                       unsigned short* __restrict__ dst, long n) {
    long i = (long)blockIdx.x * blockDim.x + threadIdx.x;
    long stride = (long)gridDim.x * blockDim.x;
    for (; i < n; i += stride) dst[i] = f2bf(src[i]);
}

// f32 [SR x SC] -> bf16 transposed [SC x SR]; writes coalesced
__global__ __launch_bounds__(256) void cvt_bf16_t_kernel(const float* __restrict__ src,
                                                         unsigned short* __restrict__ dst,
                                                         int SR, int SC) {
    long n = (long)SR * SC;
    long stride = (long)gridDim.x * blockDim.x;
    for (long i = (long)blockIdx.x * blockDim.x + threadIdx.x; i < n; i += stride) {
        long c = i / SR, r = i % SR;
        dst[i] = f2bf(src[r * (long)SC + c]);
    }
}

// ---------------------------------------------------------------------------
// LDS-free WMMA GEMM: C[16 x 768] per block, 8 waves x (16x96), K-step 32.
//   A  : bf16 row-major [rows x K]   (batched via aBatchStride/rowsPerBatch)
//   BT : bf16 transposed [768 x K]   (row stride btRowStride, batched)
//   DOLN   : fused bias + LayerNorm(768) + ReLU epilogue
//   OUTF32 : LN result written f32 (else bf16)
//   OUTT   : additionally write bf16 transposed copy (packed 16B chunks)
// ---------------------------------------------------------------------------
template <bool DOLN, bool OUTF32, bool OUTT>
__global__ __launch_bounds__(256) void gemm_bf16_kernel(
    const unsigned short* __restrict__ A, long aRowStride, long aBatchStride, int rowsPerBatch,
    const unsigned short* __restrict__ BT, long btRowStride, long btBatchStride, int K,
    const float* __restrict__ bias, const float* __restrict__ gamma, const float* __restrict__ beta,
    float* __restrict__ outF, unsigned short* __restrict__ outB, int ldOut, int colOff,
    unsigned short* __restrict__ outT, long ldT)
{
    __shared__ float lRedS[DOLN ? 2048 : 1];
    __shared__ float lRedQ[DOLN ? 2048 : 1];
    __shared__ float l2S[DOLN ? 256 : 1];
    __shared__ float l2Q[DOLN ? 256 : 1];
    __shared__ float lMean[16], lRstd[16];

    const int tid   = threadIdx.x;
    const int wave  = tid >> 5;
    const int lane  = tid & 31;
    const int lr    = lane & 15;
    const int khalf = lane >> 4;
    const int r0    = blockIdx.x * 16;
    const int batch = r0 / rowsPerBatch;
    const int rloc  = r0 % rowsPerBatch;

    const unsigned short* arow = A + (size_t)batch * aBatchStride
                                   + (size_t)(rloc + lr) * aRowStride + 16 * khalf;
    const unsigned short* brow[6];
#pragma unroll
    for (int nt = 0; nt < 6; ++nt) {
        int col = wave * 96 + nt * 16 + lr;
        brow[nt] = BT + (size_t)batch * btBatchStride + (size_t)col * btRowStride + 16 * khalf;
    }

    v8f acc[6];
    v8f zero = {0.f, 0.f, 0.f, 0.f, 0.f, 0.f, 0.f, 0.f};
#pragma unroll
    for (int i = 0; i < 6; ++i) acc[i] = zero;

    for (int k0 = 0; k0 < K; k0 += 32) {
        v16bf af = a_frag(arow + k0);
        if (k0 + 64 < K) {
            __builtin_prefetch(arow + k0 + 64, 0, 0);
            __builtin_prefetch(brow[0] + k0 + 64, 0, 0);
        }
#pragma unroll
        for (int nt = 0; nt < 6; ++nt) {
            v16bf bf = b_frag(brow[nt] + k0);
            acc[nt] = wmma_bf16(af, bf, acc[nt]);
        }
    }

    if constexpr (DOLN) {
        // bias add + per-row sum/sumsq from registers; cross-wave reduce in LDS
#pragma unroll
        for (int j = 0; j < 8; ++j) {
            float s = 0.f, q = 0.f;
#pragma unroll
            for (int nt = 0; nt < 6; ++nt) {
                int col = wave * 96 + nt * 16 + lr;
                float x = acc[nt][j] + bias[col];
                acc[nt][j] = x;
                s += x; q += x * x;
            }
            int row = j + 8 * khalf;
            lRedS[row * 128 + wave * 16 + lr] = s;
            lRedQ[row * 128 + wave * 16 + lr] = q;
        }
        __syncthreads();
        int r = tid >> 4, sub = tid & 15;
        {
            float s = 0.f, q = 0.f;
#pragma unroll
            for (int i = 0; i < 8; ++i) {
                s += lRedS[r * 128 + sub + 16 * i];
                q += lRedQ[r * 128 + sub + 16 * i];
            }
            l2S[tid] = s; l2Q[tid] = q;
        }
        __syncthreads();
        if (sub == 0) {
            float ts = 0.f, tq = 0.f;
            for (int i = 0; i < 16; ++i) { ts += l2S[r * 16 + i]; tq += l2Q[r * 16 + i]; }
            float mean = ts * (1.f / 768.f);
            float var  = tq * (1.f / 768.f) - mean * mean;
            lMean[r] = mean;
            lRstd[r] = rsqrtf(var + 1e-5f);
        }
        __syncthreads();
#pragma unroll
        for (int nt = 0; nt < 6; ++nt) {
            int col = wave * 96 + nt * 16 + lr;
            float g = gamma[col], be = beta[col];
            unsigned int pk[4] = {0, 0, 0, 0};
#pragma unroll
            for (int j = 0; j < 8; ++j) {
                int row = j + 8 * khalf;
                float y = (acc[nt][j] - lMean[row]) * lRstd[row] * g + be;
                y = fmaxf(y, 0.f);
                size_t orow = (size_t)(r0 + row);
                if (OUTF32) outF[orow * 768 + col] = y;
                else        outB[orow * 768 + col] = f2bf(y);
                if (OUTT) {
                    unsigned int h = f2bf(y);
                    pk[j >> 1] |= h << ((j & 1) * 16);
                }
            }
            if (OUTT) {
                uint4 v; v.x = pk[0]; v.y = pk[1]; v.z = pk[2]; v.w = pk[3];
                *(uint4*)(outT + (size_t)col * ldT + r0 + 8 * khalf) = v;
            }
        }
    } else {
        // plain bf16 store (attention-apply outputs into concat buffer)
#pragma unroll
        for (int j = 0; j < 8; ++j) {
            int row = j + 8 * khalf;
#pragma unroll
            for (int nt = 0; nt < 6; ++nt) {
                int col = wave * 96 + nt * 16 + lr;
                outB[(size_t)(r0 + row) * ldOut + colOff + col] = f2bf(acc[nt][j]);
            }
        }
    }
}

// ---------------------------------------------------------------------------
// sim = vp @ lp^T * scale, softmax over 2048 cols; writes A (bf16, row-major)
// and A^T (bf16, packed 16B chunks). Scores live in 128 accumulator VGPRs.
// ---------------------------------------------------------------------------
__global__ __launch_bounds__(256) void sim_softmax_kernel(
    const unsigned short* __restrict__ vpb, const unsigned short* __restrict__ lpb,
    const float* __restrict__ scalep,
    unsigned short* __restrict__ Aout, unsigned short* __restrict__ ATout)
{
    __shared__ float lRed[16 * 128];
    __shared__ float lRed2[256];
    __shared__ float lRow[16];

    const int tid   = threadIdx.x;
    const int wave  = tid >> 5;
    const int lane  = tid & 31;
    const int lr    = lane & 15;
    const int khalf = lane >> 4;
    const int b     = blockIdx.x >> 7;          // 128 row-tiles per batch
    const int s0    = (blockIdx.x & 127) << 4;

    const unsigned short* vbp = vpb + (size_t)b * 2048 * 768;
    const unsigned short* lbp = lpb + (size_t)b * 2048 * 768;
    const unsigned short* arow = vbp + (size_t)(s0 + lr) * 768 + 16 * khalf;
    const unsigned short* brow[16];
#pragma unroll
    for (int nt = 0; nt < 16; ++nt) {
        int t = wave * 256 + nt * 16 + lr;
        brow[nt] = lbp + (size_t)t * 768 + 16 * khalf;
    }

    v8f acc[16];
    v8f zero = {0.f, 0.f, 0.f, 0.f, 0.f, 0.f, 0.f, 0.f};
#pragma unroll
    for (int i = 0; i < 16; ++i) acc[i] = zero;

    for (int k0 = 0; k0 < 768; k0 += 32) {
        v16bf af = a_frag(arow + k0);
        if (k0 + 64 < 768) __builtin_prefetch(arow + k0 + 64, 0, 0);
#pragma unroll
        for (int nt = 0; nt < 16; ++nt) {
            v16bf bf = b_frag(brow[nt] + k0);
            acc[nt] = wmma_bf16(af, bf, acc[nt]);
        }
    }

    const float sc = *scalep;
#pragma unroll
    for (int nt = 0; nt < 16; ++nt)
#pragma unroll
        for (int j = 0; j < 8; ++j) acc[nt][j] *= sc;

    // ---- row max ----
#pragma unroll
    for (int j = 0; j < 8; ++j) {
        float m = -3.4e38f;
#pragma unroll
        for (int nt = 0; nt < 16; ++nt) m = fmaxf(m, acc[nt][j]);
        lRed[(j + 8 * khalf) * 128 + wave * 16 + lr] = m;
    }
    __syncthreads();
    {
        int r = tid >> 4, sub = tid & 15;
        float m = -3.4e38f;
#pragma unroll
        for (int i = 0; i < 8; ++i) m = fmaxf(m, lRed[r * 128 + sub + 16 * i]);
        lRed2[tid] = m;
        __syncthreads();
        if (sub == 0) {
            float mm = lRed2[r * 16];
            for (int i = 1; i < 16; ++i) mm = fmaxf(mm, lRed2[r * 16 + i]);
            lRow[r] = mm;
        }
    }
    __syncthreads();

    // ---- exp + row sum ----
#pragma unroll
    for (int j = 0; j < 8; ++j) {
        float rowm = lRow[j + 8 * khalf];
        float s = 0.f;
#pragma unroll
        for (int nt = 0; nt < 16; ++nt) {
            float e = __expf(acc[nt][j] - rowm);
            acc[nt][j] = e;
            s += e;
        }
        lRed[(j + 8 * khalf) * 128 + wave * 16 + lr] = s;
    }
    __syncthreads();
    {
        int r = tid >> 4, sub = tid & 15;
        float s = 0.f;
#pragma unroll
        for (int i = 0; i < 8; ++i) s += lRed[r * 128 + sub + 16 * i];
        lRed2[tid] = s;
        __syncthreads();
        if (sub == 0) {
            float ss = 0.f;
            for (int i = 0; i < 16; ++i) ss += lRed2[r * 16 + i];
            lRow[r] = ss;
        }
    }
    __syncthreads();

    // ---- normalize + store A (row-major) and A^T (packed 16B chunks) ----
    float inv[8];
#pragma unroll
    for (int j = 0; j < 8; ++j) inv[j] = 1.f / lRow[j + 8 * khalf];

    unsigned short* Ab  = Aout  + (size_t)b * 2048 * 2048;
    unsigned short* ATb = ATout + (size_t)b * 2048 * 2048;
#pragma unroll
    for (int nt = 0; nt < 16; ++nt) {
        int col = wave * 256 + nt * 16 + lr;
        unsigned int pk[4] = {0, 0, 0, 0};
#pragma unroll
        for (int j = 0; j < 8; ++j) {
            int row = j + 8 * khalf;
            unsigned int h = f2bf(acc[nt][j] * inv[j]);
            Ab[(size_t)(s0 + row) * 2048 + col] = (unsigned short)h;
            pk[j >> 1] |= h << ((j & 1) * 16);
        }
        uint4 v; v.x = pk[0]; v.y = pk[1]; v.z = pk[2]; v.w = pk[3];
        *(uint4*)(ATb + (size_t)col * 2048 + s0 + 8 * khalf) = v;
    }
}

// ---------------------------------------------------------------------------
// Host-side launcher
// ---------------------------------------------------------------------------
extern "C" void kernel_launch(void* const* d_in, const int* in_sizes, int n_in,
                              void* d_out, int out_size, void* d_ws, size_t ws_size,
                              hipStream_t stream) {
    (void)in_sizes; (void)n_in; (void)out_size; (void)ws_size;
    const float* vis   = (const float*)d_in[0];
    const float* lang  = (const float*)d_in[1];
    const float* vW    = (const float*)d_in[2];
    const float* vb    = (const float*)d_in[3];
    const float* vg    = (const float*)d_in[4];
    const float* vbeta = (const float*)d_in[5];
    const float* lW    = (const float*)d_in[6];
    const float* lb    = (const float*)d_in[7];
    const float* lg    = (const float*)d_in[8];
    const float* lbeta = (const float*)d_in[9];
    const float* claw  = (const float*)d_in[10];
    const float* oW    = (const float*)d_in[11];
    const float* ob    = (const float*)d_in[12];
    const float* og    = (const float*)d_in[13];
    const float* obeta = (const float*)d_in[14];
    float* out = (float*)d_out;

    const int  BS = 8 * 2048;             // 16384 flattened rows
    const long SS = 2048L * 2048L;

    char* ws = (char*)d_ws;
    size_t off = 0;
    auto alloc = [&](size_t bytes) { size_t o = off; off += (bytes + 255) & ~(size_t)255; return o; };
    float*          scale = (float*)(ws + alloc(4));
    unsigned short* visb  = (unsigned short*)(ws + alloc((size_t)BS * 768 * 2));
    unsigned short* langb = (unsigned short*)(ws + alloc((size_t)BS * 768 * 2));
    unsigned short* WvT   = (unsigned short*)(ws + alloc((size_t)768 * 768 * 2));
    unsigned short* WlT   = (unsigned short*)(ws + alloc((size_t)768 * 768 * 2));
    unsigned short* WoT   = (unsigned short*)(ws + alloc((size_t)1536 * 768 * 2));
    unsigned short* vpb   = (unsigned short*)(ws + alloc((size_t)BS * 768 * 2));
    unsigned short* vpbT  = (unsigned short*)(ws + alloc((size_t)BS * 768 * 2));
    unsigned short* lpb   = (unsigned short*)(ws + alloc((size_t)BS * 768 * 2));
    unsigned short* lpbT  = (unsigned short*)(ws + alloc((size_t)BS * 768 * 2));
    unsigned short* Amat  = (unsigned short*)(ws + alloc((size_t)8 * SS * 2));
    unsigned short* AmatT = (unsigned short*)(ws + alloc((size_t)8 * SS * 2));
    unsigned short* comb  = (unsigned short*)(ws + alloc((size_t)BS * 1536 * 2));

    // 0) claw mean / temperature
    claw_scale_kernel<<<1, 1024, 0, stream>>>(claw, scale);
    // 1) activations -> bf16 row-major; weights -> bf16 transposed [N][K]
    cvt_bf16_kernel<<<2048, 256, 0, stream>>>(vis,  visb,  (long)BS * 768);
    cvt_bf16_kernel<<<2048, 256, 0, stream>>>(lang, langb, (long)BS * 768);
    cvt_bf16_t_kernel<<<512, 256, 0, stream>>>(vW, WvT, 768, 768);
    cvt_bf16_t_kernel<<<512, 256, 0, stream>>>(lW, WlT, 768, 768);
    cvt_bf16_t_kernel<<<512, 256, 0, stream>>>(oW, WoT, 1536, 768);
    // 2) projections: X@W + b, LN, ReLU -> vp/lp (row-major + transposed)
    gemm_bf16_kernel<true, false, true><<<BS / 16, 256, 0, stream>>>(
        visb, 768, 0, BS, WvT, 768, 0, 768,
        vb, vg, vbeta, nullptr, vpb, 768, 0, vpbT, BS);
    gemm_bf16_kernel<true, false, true><<<BS / 16, 256, 0, stream>>>(
        langb, 768, 0, BS, WlT, 768, 0, 768,
        lb, lg, lbeta, nullptr, lpb, 768, 0, lpbT, BS);
    // 3) sim + softmax -> A and A^T (bf16)
    sim_softmax_kernel<<<8 * 128, 256, 0, stream>>>(vpb, lpb, scale, Amat, AmatT);
    // 4) aligned_vision = A @ vp  -> comb[:, 0:768]
    gemm_bf16_kernel<false, false, false><<<BS / 16, 256, 0, stream>>>(
        Amat, 2048, SS, 2048, vpbT, BS, 2048, 2048,
        nullptr, nullptr, nullptr, nullptr, comb, 1536, 0, nullptr, 0);
    // 5) aligned_language = A^T @ lp -> comb[:, 768:1536]
    gemm_bf16_kernel<false, false, false><<<BS / 16, 256, 0, stream>>>(
        AmatT, 2048, SS, 2048, lpbT, BS, 2048, 2048,
        nullptr, nullptr, nullptr, nullptr, comb, 1536, 768, nullptr, 0);
    // 6) output: comb @ oW + ob, LN, ReLU -> f32 d_out
    gemm_bf16_kernel<true, true, false><<<BS / 16, 256, 0, stream>>>(
        comb, 1536, 0, BS, WoT, 1536, 0, 1536,
        ob, og, obeta, out, nullptr, 768, 0, nullptr, 0);
}
